// DGINConv_39728447488463
// MI455X (gfx1250) — matrix operand order repaired and checked by reading, not compiled, not run on hardware
//
#include <hip/hip_runtime.h>

typedef __attribute__((ext_vector_type(16))) _Float16 v16h;
typedef __attribute__((ext_vector_type(8)))  float    v8f;
typedef __attribute__((ext_vector_type(2)))  float    v2f;
typedef __attribute__((ext_vector_type(4)))  float    v4f;

#define Nn   1024
#define Dd   128
#define Ee   32
#define WROW 160   // D + E columns of Wne
#define MAXN 128   // neighbor-list capacity (mean ~31, P(>128) ~ 0)
#define EPAD 36    // padded row stride (dwords) for LDS edge tile (bank spread, 16B aligned)

// ---------------------------------------------------------------------------
// Build neighbor lists from dense adjacency; deterministic sequential scan,
// PADDED to a multiple of 16 with (j=0, w=0) so the WMMA loop is branch-free.
// ---------------------------------------------------------------------------
__global__ __launch_bounds__(256) void build_nbr_kernel(
    const float* __restrict__ adj, int* __restrict__ nbr_idx,
    float* __restrict__ nbr_w, int* __restrict__ nbr_cnt)
{
    const int i = blockIdx.x * blockDim.x + threadIdx.x;
    if (i >= Nn) return;
    int c = 0;
    const float* row = adj + (size_t)i * Nn;
    for (int j = 0; j < Nn; ++j) {
        const float a = row[j];
        if (a != 0.0f && c < MAXN) {
            nbr_idx[(size_t)i * MAXN + c] = j;
            nbr_w  [(size_t)i * MAXN + c] = a;
            ++c;
        }
    }
    nbr_cnt[i] = c;
    int cpad = (c + 15) & ~15;
    for (; c < cpad; ++c) {                 // zero-weight padding -> no guards needed
        nbr_idx[(size_t)i * MAXN + c] = 0;
        nbr_w  [(size_t)i * MAXN + c] = 0.0f;
    }
}

// ---------------------------------------------------------------------------
// hnb[j][d] = x[j,:] . Wne[d,0:128] + bne[d]  via full-precision f32 WMMA:
// v_wmma_f32_16x16x4_f32, K=128 chained in 32 steps of K=4.
// ---------------------------------------------------------------------------
__global__ __launch_bounds__(256) void hnode_kernel(
    const float* __restrict__ x, const float* __restrict__ Wne,
    const float* __restrict__ bne, float* __restrict__ hnb)
{
    const int i0   = blockIdx.x * 16;
    const int wave = threadIdx.x >> 5;
    const int lane = threadIdx.x & 31;
    const int hi   = lane >> 4;
    const int ln   = lane & 15;
    const int dN   = wave * 16 + ln;

    v8f c = {};
    const float* xrow = x   + (size_t)(i0 + ln) * Dd   + 2 * hi; // A row m=ln, K pair
    const float* wrow = Wne + (size_t)dN * WROW        + 2 * hi; // B col n=ln (node part)
    #pragma unroll 4
    for (int k0 = 0; k0 < Dd; k0 += 4) {
        v2f a = *(const v2f*)(xrow + k0);
        v2f b = *(const v2f*)(wrow + k0);
        c = __builtin_amdgcn_wmma_f32_16x16x4_f32(false, a, false, b,
                                                  (short)0, c, false, false);
    }
    const float bv = bne[dN];
    #pragma unroll
    for (int r = 0; r < 8; ++r)
        hnb[(size_t)(i0 + r + 8 * hi) * Dd + dN] = c[r] + bv;
}

// ---------------------------------------------------------------------------
// Fused GIN layer, one block per node i, 8 waves = 8 feature tiles of 16.
// Double-buffered async staging of edge rows (global_load_async_to_lds_b128,
// ASYNCcnt) overlapped with WMMA + gather epilogue; one barrier per group.
// ---------------------------------------------------------------------------
__global__ __launch_bounds__(256) void gin_layer_kernel(
    const float* __restrict__ x,        // (N,128) layer input
    const float* __restrict__ edges,    // (N,N,32)
    const float* __restrict__ Wne,      // (128,160); edge-part cols [128,160)
    const float* __restrict__ hnb,      // (N,128) = x@Wnode^T + bne
    const int*   __restrict__ nbr_idx,  // (N,MAXN) padded
    const float* __restrict__ nbr_w,    // (N,MAXN) padded
    const int*   __restrict__ nbr_cnt,  // (N)
    const float* __restrict__ eps,      // (1)
    const float* __restrict__ Wn,       // (128,128)
    const float* __restrict__ bn,       // (128)
    float* __restrict__ out)            // (N,128)
{
    const int i    = blockIdx.x;
    const int tid  = threadIdx.x;
    const int wave = tid >> 5;
    const int lane = tid & 31;
    const int hi   = lane >> 4;
    const int ln   = lane & 15;

    __shared__ __align__(16) float ebuf[2][16 * EPAD]; // double-buffered edge tiles
    __shared__ int   sidx[MAXN];
    __shared__ float sw  [MAXN];
    __shared__ float row [Dd];

    // --- stage padded neighbor list (indices + weights) once per block
    if (tid < MAXN) {
        sidx[tid] = nbr_idx[(size_t)i * MAXN + tid];
        sw  [tid] = nbr_w  [(size_t)i * MAXN + tid];
    }

    // --- B tile (f16, contiguous K order): lane covers N=ln, K = hi*16 .. hi*16+15
    const int d0 = wave * 16;
    const int dN = d0 + ln;
    v16h bt;
    {
        const float* wp = Wne + (size_t)dN * WROW + Dd + hi * 16;
        #pragma unroll
        for (int q = 0; q < 16; ++q) bt[q] = (_Float16)wp[q];
    }

    const int cnt     = nbr_cnt[i];
    const int ngroups = (cnt + 15) >> 4;
    const int slot    = tid >> 3;            // staging: neighbor slot (tid<128)
    const int part    = tid & 7;             // staging: 16B chunk of 128B row
    float acc = 0.0f;                        // partial msg[i, dN]

    __syncthreads();                         // sidx/sw visible to all waves

    // --- prologue: stage group 0 into buffer 0
    if (ngroups > 0 && tid < 128) {
        const int j = sidx[slot];
        const float* src = edges + ((size_t)i * Nn + j) * Ee + part * 4;
        unsigned ldsoff = (unsigned)(size_t)(&ebuf[0][slot * EPAD + part * 4]);
        asm volatile("global_load_async_to_lds_b128 %0, %1, off"
                     :: "v"(ldsoff), "v"(src) : "memory");
    }

    for (int g = 0; g < ngroups; ++g) {
        const int buf = g & 1;

        // --- epilogue gathers for group g: depend only on sidx, overlap the
        // in-flight async edge staging (loadcnt waits land at the consume).
        int jj[8]; float ww[8]; float hv[8];
        #pragma unroll
        for (int r = 0; r < 8; ++r) {
            const int s2 = g * 16 + r + 8 * hi;
            jj[r] = sidx[s2];
            ww[r] = sw[s2];
        }
        #pragma unroll
        for (int r = 0; r < 8; ++r)          // 8 independent, clause-able gathers
            hv[r] = hnb[(size_t)jj[r] * Dd + dN];

        // group g staged (next group not yet issued -> asynccnt 0 covers g only)
        asm volatile("s_wait_asynccnt 0" ::: "memory");
        __syncthreads();                     // all waves: ebuf[buf] ready AND
                                             // everyone finished reading ebuf[buf^1]
        // --- issue group g+1 into the alternate buffer; completes during compute
        if (g + 1 < ngroups && tid < 128) {
            const int j = sidx[(g + 1) * 16 + slot];
            const float* src = edges + ((size_t)i * Nn + j) * Ee + part * 4;
            unsigned ldsoff = (unsigned)(size_t)(&ebuf[buf ^ 1][slot * EPAD + part * 4]);
            asm volatile("global_load_async_to_lds_b128 %0, %1, off"
                         :: "v"(ldsoff), "v"(src) : "memory");
        }

        // --- A fragment from LDS (f16): row M=ln;
        // lanes 0-15: K 0..15 ; lanes 16-31: K 16..31
        const float* er = &ebuf[buf][ln * EPAD + hi * 16];
        v4f f0 = *(const v4f*)(er);
        v4f f1 = *(const v4f*)(er + 4);
        v4f f2 = *(const v4f*)(er + 8);
        v4f f3 = *(const v4f*)(er + 12);
        v16h at;
        #pragma unroll
        for (int q = 0; q < 4; ++q) {
            at[q]      = (_Float16)f0[q];
            at[4 + q]  = (_Float16)f1[q];
            at[8 + q]  = (_Float16)f2[q];
            at[12 + q] = (_Float16)f3[q];
        }

        v8f c = {};
        c = __builtin_amdgcn_wmma_f32_16x16x32_f16(
                false, at, false, bt, (short)0, c, false, false);

        // c[r] = he[i, j_(g*16+r+8*hi), dN]; branch-free (padding has w=0)
        #pragma unroll
        for (int r = 0; r < 8; ++r) {
            float v = c[r] + hv[r];
            v = v > 0.0f ? v : 0.0f;
            acc += ww[r] * v;
        }
    }

    // lanes l and l+16 hold the same feature dN -> fold halves
    acc += __shfl_xor(acc, 16, 32);
    if (hi == 0)
        row[dN] = (1.0f + eps[0]) * x[(size_t)i * Dd + dN] + acc;
    __syncthreads();

    // --- fused output GEMM: out[i,k] = relu(row . Wn[k,:] + bn[k])
    if (tid < Dd) {
        const int k = tid;
        float s = bn[k];
        const float4* wr4 = (const float4*)(Wn + (size_t)k * Dd);
        const float4* rr4 = (const float4*)row;
        #pragma unroll 8
        for (int q = 0; q < Dd / 4; ++q) {
            float4 w4 = wr4[q], r4 = rr4[q];
            s += w4.x * r4.x + w4.y * r4.y + w4.z * r4.z + w4.w * r4.w;
        }
        out[(size_t)i * Dd + k] = s > 0.0f ? s : 0.0f;
    }
}

// ---------------------------------------------------------------------------
extern "C" void kernel_launch(void* const* d_in, const int* in_sizes, int n_in,
                              void* d_out, int out_size, void* d_ws, size_t ws_size,
                              hipStream_t stream) {
    const float* adj   = (const float*)d_in[0];
    const float* nodes = (const float*)d_in[1];
    const float* edges = (const float*)d_in[2];
    const float* eps   = (const float*)d_in[3];
    const float* Wne0  = (const float*)d_in[4];
    const float* bne0  = (const float*)d_in[5];
    const float* Wn0   = (const float*)d_in[6];
    const float* bn0   = (const float*)d_in[7];
    const float* Wne1  = (const float*)d_in[8];
    const float* bne1  = (const float*)d_in[9];
    const float* Wn1   = (const float*)d_in[10];
    const float* bn1   = (const float*)d_in[11];

    char* ws = (char*)d_ws;
    float* hnb     = (float*)ws;  ws += (size_t)Nn * Dd   * sizeof(float);
    float* h1      = (float*)ws;  ws += (size_t)Nn * Dd   * sizeof(float);
    int*   nbr_idx = (int*)  ws;  ws += (size_t)Nn * MAXN * sizeof(int);
    float* nbr_w   = (float*)ws;  ws += (size_t)Nn * MAXN * sizeof(float);
    int*   nbr_cnt = (int*)  ws;

    float* outp = (float*)d_out;

    build_nbr_kernel<<<(Nn + 255) / 256, 256, 0, stream>>>(adj, nbr_idx, nbr_w, nbr_cnt);

    // ---- layer 0
    hnode_kernel<<<Nn / 16, 256, 0, stream>>>(nodes, Wne0, bne0, hnb);
    gin_layer_kernel<<<Nn, 256, 0, stream>>>(nodes, edges, Wne0, hnb,
                                             nbr_idx, nbr_w, nbr_cnt,
                                             eps, Wn0, bn0, h1);
    // ---- layer 1
    hnode_kernel<<<Nn / 16, 256, 0, stream>>>(h1, Wne1, bne1, hnb);
    gin_layer_kernel<<<Nn, 256, 0, stream>>>(h1, edges, Wne1, hnb,
                                             nbr_idx, nbr_w, nbr_cnt,
                                             eps, Wn1, bn1, outp);
}
